// textGRU_90915867721770
// MI455X (gfx1250) — matrix operand order repaired and checked by reading, not compiled, not run on hardware
//
#include <hip/hip_runtime.h>
#include <stdint.h>

typedef __attribute__((ext_vector_type(16))) __bf16 bf16x16;
typedef __attribute__((ext_vector_type(8)))  float  f32x8;

#define BB 64     // batch
#define SS 512    // seq len
#define EE 512    // embed dim
#define HD 1024   // hidden
#define OO 10     // output classes
#define NX 3072   // 3*HD (r|z|c)

// ---------- scalar helpers ----------
__device__ __forceinline__ __bf16 f2bf(float f) {
  union { float f; uint32_t u; } x; x.f = f;
  uint32_t r = x.u + 0x7FFFu + ((x.u >> 16) & 1u);   // round-to-nearest-even
  union { uint16_t u; __bf16 b; } y; y.u = (uint16_t)(r >> 16);
  return y.b;
}
__device__ __forceinline__ float bf2f(__bf16 b) {
  union { uint16_t u; __bf16 b; } y; y.b = b;
  union { uint32_t u; float f; } x; x.u = ((uint32_t)y.u) << 16;
  return x.f;
}

// ---------- WMMA fragment ----------
union Frag { bf16x16 v; uint4 u[2]; };

// Block = 256 threads (8 waves). Block tile = 64M x 64N, K step = 32.
// Wave w: rows [(w&3)*16, +16), cols [(w>>2)*32, +32)  -> 2 WMMAs per k-step.
// B pre-packed: element (k,n) at Bp[(k/32)*N*32 + n*32 + (k%32)]; the 64-col
// panel (4KB) is staged through double-buffered LDS so the kb+1 global->LDS
// copy overlaps the kb WMMAs; one barrier per k-step.
__device__ __forceinline__ void gemm_core(
    const __bf16* __restrict__ A, int lda, int row_base,
    const __bf16* __restrict__ Bp, int N, int n0, int K,
    f32x8 acc[2])
{
  __shared__ __bf16 ldsB[2][64 * 32];
  const int tid  = threadIdx.x;
  const int lane = tid & 31;
  const int wave = tid >> 5;
  const int l    = lane & 15;
  const int hi   = lane >> 4;
  const int rowStrip = wave & 3;
  const int colHalf  = wave >> 2;
  const int row  = row_base + rowStrip * 16 + l;

#pragma unroll
  for (int c = 0; c < 2; ++c)
#pragma unroll
    for (int i = 0; i < 8; ++i) acc[c][i] = 0.0f;

  const int stgCol = tid >> 2;       // 64 cols, one uint4 (16B) per thread
  const int stgQtr = tid & 3;
  const int nkb = K >> 5;

  // prologue: stage panel 0
  {
    const uint4* g = (const uint4*)(Bp + ((size_t)(n0 + stgCol)) * 32) + stgQtr;
    *((uint4*)(ldsB[0] + stgCol * 32) + stgQtr) = *g;
  }
  __syncthreads();

  for (int kb = 0; kb < nkb; ++kb) {
    const int cur = kb & 1;
    if (kb + 1 < nkb) {  // stage next panel into the other buffer (overlaps WMMA)
      const uint4* g =
          (const uint4*)(Bp + ((size_t)(kb + 1) * N + (size_t)(n0 + stgCol)) * 32) + stgQtr;
      *((uint4*)(ldsB[cur ^ 1] + stgCol * 32) + stgQtr) = *g;
    }

    // A fragment: 16x32 bf16 per the ISA lane layout (two b128 loads)
    Frag a;
    const uint4* ap = (const uint4*)(A + (size_t)row * lda + (size_t)kb * 32);
    a.u[0] = ap[hi];        // lanes<16: k0..7  ; lanes>=16: k8..15
    a.u[1] = ap[hi + 2];    // lanes<16: k16..23; lanes>=16: k24..31

#pragma unroll
    for (int c = 0; c < 2; ++c) {
      Frag b;
      const uint4* sp =
          (const uint4*)(ldsB[cur] + (colHalf * 32 + c * 16 + l) * 32 + hi * 8);
      b.u[0] = sp[0];
      b.u[1] = sp[2];
      acc[c] = __builtin_amdgcn_wmma_f32_16x16x32_bf16(
          false, a.v, false, b.v, (short)0, acc[c], false, false);
    }
    __syncthreads();   // panel cur fully read; panel cur^1 fully staged
  }
}

// ---------- prep kernels ----------
// Gather embeddings to time-major bf16: Xemb[(s*64+b)][e] = bf16(Wemb[tok[b][s]][e])
__global__ __launch_bounds__(256) void k_gather(const int* __restrict__ tok,
                                                const float* __restrict__ Wemb,
                                                __bf16* __restrict__ Xemb)
{
  int g = blockIdx.x * 256 + threadIdx.x;          // one 8-elem chunk per thread
  if (g >= BB * SS * (EE / 8)) return;
  int m  = g >> 6;                                  // EE/8 = 64 chunks per row
  int e0 = (g & 63) << 3;
  int s = m >> 6, b = m & 63;
  int t = tok[b * SS + s];
  const float* src = Wemb + (size_t)t * EE + e0;
  union { __bf16 h[8]; uint4 u; } tmp;
#pragma unroll
  for (int i = 0; i < 8; ++i) tmp.h[i] = f2bf(src[i]);
  *(uint4*)(Xemb + (size_t)m * EE + e0) = tmp.u;
}

// Pack fp32 [K x 1024] sources (up to 3 concatenated along N) into WMMA-B layout
__global__ __launch_bounds__(256) void k_pack(const float* __restrict__ s0,
                                              const float* __restrict__ s1,
                                              const float* __restrict__ s2,
                                              __bf16* __restrict__ dst, int K, int N)
{
  int e = blockIdx.x * 256 + threadIdx.x;
  if (e >= K * N) return;
  int kb  = e / (N * 32);
  int rem = e - kb * (N * 32);
  int n   = rem >> 5;
  int k   = (kb << 5) + (rem & 31);
  int sel = n >> 10;
  int nn  = n & 1023;
  const float* src = (sel == 0) ? s0 : (sel == 1) ? s1 : s2;
  dst[e] = f2bf(src[(size_t)k * HD + nn]);
}

__global__ __launch_bounds__(256) void k_hinit(float* Hf, __bf16* Hb)
{
  int i = blockIdx.x * 256 + threadIdx.x;
  if (i < BB * HD) { Hf[i] = 0.0f; Hb[i] = f2bf(0.0f); }
}

// ---------- big parallel GEMM: Xproj = Xemb @ [Wxr|Wxz|Wxc] + bias ----------
__global__ __launch_bounds__(256) void k_xproj(
    const __bf16* __restrict__ Xemb, const __bf16* __restrict__ WxP,
    const float* __restrict__ b_r, const float* __restrict__ b_z,
    const float* __restrict__ b_c, __bf16* __restrict__ Xproj)
{
  const int n0 = blockIdx.x * 64;
  const int m0 = blockIdx.y * 64;
  f32x8 acc[2];
  gemm_core(Xemb, EE, m0, WxP, NX, n0, EE, acc);
  const int lane = threadIdx.x & 31, wave = threadIdx.x >> 5;
  const int l = lane & 15, hi = lane >> 4;
  const int rowStrip = wave & 3, colHalf = wave >> 2;
#pragma unroll
  for (int c = 0; c < 2; ++c) {
    const int n = n0 + colHalf * 32 + c * 16 + l;
    const float bias = (n < HD) ? b_r[n] : (n < 2 * HD) ? b_z[n - HD] : b_c[n - 2 * HD];
#pragma unroll
    for (int r = 0; r < 8; ++r) {
      const int m = m0 + rowStrip * 16 + hi * 8 + r;
      Xproj[(size_t)m * NX + n] = f2bf(acc[c][r] + bias);
    }
  }
}

// ---------- recurrent step, phase 1: r,z gates ----------
__global__ __launch_bounds__(256) void k_gates(
    const __bf16* __restrict__ Hb, const __bf16* __restrict__ WhrzP,
    const __bf16* __restrict__ Xproj, const float* __restrict__ Hf,
    __bf16* __restrict__ Hr, float* __restrict__ Z, int s)
{
  const int n0 = blockIdx.x * 64;
  f32x8 acc[2];
  gemm_core(Hb, HD, 0, WhrzP, 2 * HD, n0, HD, acc);
  const int lane = threadIdx.x & 31, wave = threadIdx.x >> 5;
  const int l = lane & 15, hi = lane >> 4;
  const int rowStrip = wave & 3, colHalf = wave >> 2;
  const __bf16* Xs = Xproj + (size_t)s * BB * NX;
#pragma unroll
  for (int c = 0; c < 2; ++c) {
    const int n = n0 + colHalf * 32 + c * 16 + l;
#pragma unroll
    for (int r = 0; r < 8; ++r) {
      const int m = rowStrip * 16 + hi * 8 + r;
      float pre = acc[c][r] + bf2f(Xs[(size_t)m * NX + n]);
      float g = 1.0f / (1.0f + __expf(-pre));
      if (n < HD) Hr[m * HD + n] = f2bf(Hf[m * HD + n] * g);   // H * r (A for phase 2)
      else        Z[m * HD + (n - HD)] = g;                    // z gate
    }
  }
}

// ---------- recurrent step, phase 2: candidate + state update ----------
__global__ __launch_bounds__(256) void k_cand(
    const __bf16* __restrict__ Hr, const __bf16* __restrict__ WhcP,
    const __bf16* __restrict__ Xproj, const float* __restrict__ Z,
    float* __restrict__ Hf, __bf16* __restrict__ Hb, int s)
{
  const int n0 = blockIdx.x * 64;
  f32x8 acc[2];
  gemm_core(Hr, HD, 0, WhcP, HD, n0, HD, acc);
  const int lane = threadIdx.x & 31, wave = threadIdx.x >> 5;
  const int l = lane & 15, hi = lane >> 4;
  const int rowStrip = wave & 3, colHalf = wave >> 2;
  const __bf16* Xs = Xproj + (size_t)s * BB * NX + 2 * HD;
#pragma unroll
  for (int c = 0; c < 2; ++c) {
    const int n = n0 + colHalf * 32 + c * 16 + l;
#pragma unroll
    for (int r = 0; r < 8; ++r) {
      const int m = rowStrip * 16 + hi * 8 + r;
      float pre = acc[c][r] + bf2f(Xs[(size_t)m * NX + n]);
      float h = tanhf(pre);
      float z = Z[m * HD + n];
      float hn = z * Hf[m * HD + n] + (1.0f - z) * h;
      Hf[m * HD + n] = hn;
      Hb[m * HD + n] = f2bf(hn);
    }
  }
}

// ---------- head: softmax(H @ W_hq + b_q) ----------
__global__ __launch_bounds__(32) void k_final(const float* __restrict__ Hf,
                                              const float* __restrict__ Whq,
                                              const float* __restrict__ bq,
                                              float* __restrict__ out)
{
  int row = blockIdx.x, lane = threadIdx.x;
  float acc[OO];
#pragma unroll
  for (int o = 0; o < OO; ++o) acc[o] = 0.0f;
  for (int k = lane; k < HD; k += 32) {
    float h = Hf[row * HD + k];
#pragma unroll
    for (int o = 0; o < OO; ++o) acc[o] += h * Whq[k * OO + o];
  }
#pragma unroll
  for (int o = 0; o < OO; ++o)
    for (int sh = 16; sh > 0; sh >>= 1) acc[o] += __shfl_down(acc[o], sh, 32);
  if (lane == 0) {
    float v[OO], mx = -3.0e38f, sum = 0.0f;
#pragma unroll
    for (int o = 0; o < OO; ++o) { v[o] = acc[o] + bq[o]; mx = fmaxf(mx, v[o]); }
#pragma unroll
    for (int o = 0; o < OO; ++o) { v[o] = __expf(v[o] - mx); sum += v[o]; }
    float inv = 1.0f / sum;
#pragma unroll
    for (int o = 0; o < OO; ++o) out[row * OO + o] = v[o] * inv;
  }
}

extern "C" void kernel_launch(void* const* d_in, const int* in_sizes, int n_in,
                              void* d_out, int out_size, void* d_ws, size_t ws_size,
                              hipStream_t stream)
{
  const int*   tok  = (const int*)  d_in[0];
  const float* Wemb = (const float*)d_in[1];
  const float* W_xr = (const float*)d_in[2];
  const float* W_hr = (const float*)d_in[3];
  const float* b_r  = (const float*)d_in[4];
  const float* W_xz = (const float*)d_in[5];
  const float* W_hz = (const float*)d_in[6];
  const float* b_z  = (const float*)d_in[7];
  const float* W_xc = (const float*)d_in[8];
  const float* W_hc = (const float*)d_in[9];
  const float* b_c  = (const float*)d_in[10];
  const float* W_hq = (const float*)d_in[11];
  const float* b_q  = (const float*)d_in[12];
  float* out = (float*)d_out;

  char* base = (char*)d_ws;
  size_t off = 0;
  auto alloc = [&](size_t bytes) {
    char* p = base + off;
    off += (bytes + 255) & ~(size_t)255;
    return p;
  };
  __bf16* Xemb  = (__bf16*)alloc((size_t)BB * SS * EE * 2);   //  33.6 MB
  __bf16* Xproj = (__bf16*)alloc((size_t)BB * SS * NX * 2);   // 201.3 MB
  __bf16* WxP   = (__bf16*)alloc((size_t)EE * NX * 2);        //   3.1 MB
  __bf16* WhrzP = (__bf16*)alloc((size_t)HD * 2 * HD * 2);    //   4.2 MB
  __bf16* WhcP  = (__bf16*)alloc((size_t)HD * HD * 2);        //   2.1 MB
  float*  Hf    = (float*) alloc((size_t)BB * HD * 4);
  __bf16* Hb    = (__bf16*)alloc((size_t)BB * HD * 2);
  __bf16* Hr    = (__bf16*)alloc((size_t)BB * HD * 2);
  float*  Z     = (float*) alloc((size_t)BB * HD * 4);
  if (off > ws_size) return;  // workspace too small; nothing safe to do

  // preparation (all parallel, all cheap vs. HBM bandwidth)
  k_gather<<<(BB * SS * (EE / 8) + 255) / 256, 256, 0, stream>>>(tok, Wemb, Xemb);
  k_pack<<<(EE * NX + 255) / 256, 256, 0, stream>>>(W_xr, W_xz, W_xc, WxP, EE, NX);
  k_pack<<<(HD * 2 * HD + 255) / 256, 256, 0, stream>>>(W_hr, W_hz, W_hz, WhrzP, HD, 2 * HD);
  k_pack<<<(HD * HD + 255) / 256, 256, 0, stream>>>(W_hc, W_hc, W_hc, WhcP, HD, HD);
  k_hinit<<<(BB * HD + 255) / 256, 256, 0, stream>>>(Hf, Hb);

  // hoisted input projections: [32768 x 512] @ [512 x 3072]  (WMMA)
  dim3 gx(NX / 64, (BB * SS) / 64);
  k_xproj<<<gx, 256, 0, stream>>>(Xemb, WxP, b_r, b_z, b_c, Xproj);

  // sequential GRU scan: two dependent WMMA kernels per step
  for (int s = 0; s < SS; ++s) {
    k_gates<<<(2 * HD) / 64, 256, 0, stream>>>(Hb, WhrzP, Xproj, Hf, Hr, Z, s);
    k_cand <<<HD / 64,       256, 0, stream>>>(Hr, WhcP, Xproj, Z, Hf, Hb, s);
  }

  k_final<<<BB, 32, 0, stream>>>(Hf, W_hq, b_q, out);
}
